// DeepSeekDecoderBlock_60644938219687
// MI455X (gfx1250) — matrix operand order repaired and checked by reading, not compiled, not run on hardware
//
#include <hip/hip_runtime.h>
#include <hip/hip_bf16.h>

// ---------------------------------------------------------------------------
// DeepSeek decoder block on gfx1250 (MI455X).
// bf16 WMMA (16x16x32, f32 accum) for all GEMMs; weights transposed+converted
// to bf16 once per launch so every fragment load is a contiguous b128 pair.
// Each GEMM wave computes a 32x32 C tile (4 WMMAs / 8 b128 loads per k-step)
// for register-level operand reuse. f32 VALU for RMSNorm / softmax / gating.
// Routed (gathered) MoE GEMMs with scatter-add combine.
// ---------------------------------------------------------------------------

typedef __bf16 bf16_t;
typedef __attribute__((ext_vector_type(16))) __bf16 v16bf;
typedef __attribute__((ext_vector_type(8)))  __bf16 v8bf;
typedef __attribute__((ext_vector_type(8)))  float  v8f;

constexpr int cHID = 2048, cNH = 16, cDN = 128, cDV = 128, cDQ = 192;
constexpr int cQR = 512, cKVR = 512, cE = 16, cIM = 512;
constexpr int cB = 2, cS = 1024, cT = 2048;
constexpr float cRSF = 2.5f, cEPS = 1e-6f;
constexpr float cROPE = 0.28782313662425f;  // ln(10000)/32

// ---------------- WMMA + fragment helpers ----------------------------------

__device__ __forceinline__ v8f wmma_bf16(v16bf a, v16bf b, v8f c) {
  return __builtin_amdgcn_wmma_f32_16x16x32_bf16(false, a, false, b, (short)0,
                                                 c, false, false);
}

__device__ __forceinline__ v8bf ld8(const bf16_t* p) {
  return *(const v8bf*)p;  // 16B global_load_b128 / ds_load_b128
}

__device__ __forceinline__ v16bf frag16(v8bf lo, v8bf hi) {
  v16bf r;
#pragma unroll
  for (int e = 0; e < 8; ++e) { r[e] = lo[e]; r[8 + e] = hi[e]; }
  return r;
}

// A fragment 16x32 (row-major, lda mult of 8):
// lane: sub=lane>>4, m=lane&15 ; elems 0..15 -> K=sub*16+e
__device__ __forceinline__ v16bf load_a_frag(const bf16_t* A, int lda,
                                             int row0, int k0, int lane) {
  int sub = lane >> 4, m = lane & 15;
  const bf16_t* p = A + (size_t)(row0 + m) * lda + k0 + sub * 16;
  return frag16(ld8(p), ld8(p + 8));
}

// B fragment 32x16 from TRANSPOSED storage BT[N][K] (ldk mult of 8):
// b[e] = B[k0+16*sub+e][col0+n] = BT[col0+n][k0+16*sub+e] -> contiguous
__device__ __forceinline__ v16bf load_bT_frag(const bf16_t* BT, int ldk,
                                              int k0, int col0, int lane) {
  int sub = lane >> 4, n = lane & 15;
  const bf16_t* p = BT + (size_t)(col0 + n) * ldk + k0 + 16 * sub;
  return frag16(ld8(p), ld8(p + 8));
}

// C/D: vreg r -> row = row0 + r + 8*sub ; col = col0 + n
__device__ __forceinline__ void store_c_f32(float* C, int ldc, int row0,
                                            int col0, int lane, v8f c,
                                            const float* Cin) {
  int sub = lane >> 4, n = lane & 15;
#pragma unroll
  for (int r = 0; r < 8; ++r) {
    size_t idx = (size_t)(row0 + r + 8 * sub) * ldc + col0 + n;
    float v = c[r];
    if (Cin) v += Cin[idx];
    C[idx] = v;
  }
}

// ---------------- Weight transpose + f32->bf16 convert ---------------------
// W: [z][K][N] f32  ->  WT: [z][N][K] bf16 ; LDS tile keeps both sides coalesced.

__global__ void convert_transpose(const float* __restrict__ W, bf16_t* __restrict__ WT,
                                  int K, int N) {
  __shared__ float tile[32][33];
  const float* Wz = W + (size_t)blockIdx.z * K * N;
  bf16_t* WTz = WT + (size_t)blockIdx.z * K * N;
  int kb = blockIdx.y * 32, nb = blockIdx.x * 32;
  int tx = threadIdx.x & 31, ty = threadIdx.x >> 5;  // 32 x 8
#pragma unroll
  for (int i = ty; i < 32; i += 8)
    tile[i][tx] = Wz[(size_t)(kb + i) * N + nb + tx];
  __syncthreads();
#pragma unroll
  for (int i = ty; i < 32; i += 8)
    WTz[(size_t)(nb + i) * K + kb + tx] = (bf16_t)tile[tx][i];
}

// ---------------- Generic GEMM: C(f32) = A(bf16) @ BT(bf16)^T [+ Cin] ------
// Block 256 = 8 waves (2x4), block tile 64x128; each wave 32x32 -> 4 WMMAs
// per k-step against 8 b128 loads (A/B each reused twice in registers).

__global__ void gemm_wmma(const bf16_t* __restrict__ A, const bf16_t* __restrict__ BT,
                          float* __restrict__ C, const float* __restrict__ Cin,
                          int M, int N, int K, int lda, int ldc) {
  int lane = threadIdx.x & 31;
  int wave = threadIdx.x >> 5;
  int row0 = blockIdx.y * 64 + (wave >> 2) * 32;
  int col0 = blockIdx.x * 128 + (wave & 3) * 32;
  if (row0 >= M || col0 >= N) return;
  v8f c00 = {}, c01 = {}, c10 = {}, c11 = {};
  for (int k0 = 0; k0 < K; k0 += 32) {
    if (k0 + 32 < K) {
      __builtin_prefetch(A + (size_t)(row0 + (lane & 15)) * lda + k0 + 32, 0, 1);
      __builtin_prefetch(BT + (size_t)(col0 + (lane & 15)) * K + k0 + 32, 0, 1);
    }
    v16bf a0 = load_a_frag(A, lda, row0, k0, lane);
    v16bf a1 = load_a_frag(A, lda, row0 + 16, k0, lane);
    v16bf b0 = load_bT_frag(BT, K, k0, col0, lane);
    v16bf b1 = load_bT_frag(BT, K, k0, col0 + 16, lane);
    c00 = wmma_bf16(a0, b0, c00);
    c01 = wmma_bf16(a0, b1, c01);
    c10 = wmma_bf16(a1, b0, c10);
    c11 = wmma_bf16(a1, b1, c11);
  }
  store_c_f32(C, ldc, row0, col0, lane, c00, Cin);
  store_c_f32(C, ldc, row0, col0 + 16, lane, c01, Cin);
  store_c_f32(C, ldc, row0 + 16, col0, lane, c10, Cin);
  store_c_f32(C, ldc, row0 + 16, col0 + 16, lane, c11, Cin);
}

// ---------------- RMSNorm ---------------------------------------------------

__global__ void rmsnorm_kernel(const float* __restrict__ in, const float* __restrict__ w,
                               bf16_t* __restrict__ out_bf, float* __restrict__ out_f,
                               int D, int in_stride) {
  int row = blockIdx.x;
  const float* x = in + (size_t)row * in_stride;
  __shared__ float red[8];
  float ss = 0.f;
  for (int d = threadIdx.x; d < D; d += blockDim.x) { float v = x[d]; ss = fmaf(v, v, ss); }
#pragma unroll
  for (int off = 16; off; off >>= 1) ss += __shfl_xor(ss, off, 32);
  if ((threadIdx.x & 31) == 0) red[threadIdx.x >> 5] = ss;
  __syncthreads();
  float tot = 0.f;
  int nw = blockDim.x >> 5;
  for (int i = 0; i < nw; ++i) tot += red[i];
  float scale = rsqrtf(tot / (float)D + cEPS);
  for (int d = threadIdx.x; d < D; d += blockDim.x) {
    float v = x[d] * scale * w[d];
    out_bf[(size_t)row * D + d] = (bf16_t)v;
    if (out_f) out_f[(size_t)row * D + d] = v;
  }
}

// ---------------- Q/K/V prep: reshape + RoPE -> bf16 -----------------------
// qf/kf: [B,NH,S,DQ] ; V stored TRANSPOSED: vt[B,NH,DV,S] for b128 fragments.

__device__ __forceinline__ float rope_elem(const float* src, int dr, float pos) {
  float x0 = src[dr];
  int odd = dr & 1;
  float other = src[odd ? dr - 1 : dr + 1];
  float rot = odd ? other : -other;
  float f = __expf(-(float)(dr & 31) * cROPE);
  float s, c;
  __sincosf(pos * f, &s, &c);
  return x0 * c + rot * s;
}

__global__ void qkv_prep(const float* __restrict__ q_f, const float* __restrict__ kv_f,
                         const float* __restrict__ kva_f, bf16_t* __restrict__ qf,
                         bf16_t* __restrict__ kf, bf16_t* __restrict__ vt) {
  int s = blockIdx.x, h = blockIdx.y, b = blockIdx.z;
  int tok = b * cS + s;
  size_t hb = (size_t)b * cNH + h;
  int d = threadIdx.x;  // 0..191
  float pos = (float)s;
  {
    const float* qr = q_f + (size_t)tok * (cNH * cDQ) + h * cDQ;
    float val = (d < cDN) ? qr[d] : rope_elem(qr + cDN, d - cDN, pos);
    qf[(hb * cS + s) * cDQ + d] = (bf16_t)val;
  }
  {
    const float* kr = kv_f + (size_t)tok * (cNH * (cDN + cDV)) + h * (cDN + cDV);
    float val = (d < cDN) ? kr[d]
                          : rope_elem(kva_f + (size_t)tok * (cKVR + 64) + cKVR, d - cDN, pos);
    kf[(hb * cS + s) * cDQ + d] = (bf16_t)val;
  }
  if (d < cDV)
    vt[(hb * cDV + d) * cS + s] =
        (bf16_t)kv_f[(size_t)tok * (cNH * (cDN + cDV)) + h * (cDN + cDV) + cDN + d];
}

// ---------------- Flash attention: 1 wave per 16 query rows ----------------

__global__ __launch_bounds__(32) void attn_kernel(const bf16_t* __restrict__ qf,
                                                  const bf16_t* __restrict__ kf,
                                                  const bf16_t* __restrict__ vt,
                                                  bf16_t* __restrict__ ao) {
  int lane = threadIdx.x & 31;
  int q0 = blockIdx.x * 16;
  int h = blockIdx.y, b = blockIdx.z;
  size_t hb = (size_t)b * cNH + h;
  const bf16_t* Q = qf + hb * cS * cDQ;
  const bf16_t* Kf = kf + hb * cS * cDQ;
  const bf16_t* VT = vt + hb * cDV * cS;
  int sub = lane >> 4, n = lane & 15;

  v16bf qa[6];
#pragma unroll
  for (int c = 0; c < 6; ++c) qa[c] = load_a_frag(Q, cDQ, q0, c * 32, lane);

  v8f o[8] = {};
  float rm[8], rs[8];
#pragma unroll
  for (int r = 0; r < 8; ++r) { rm[r] = -1e30f; rs[r] = 0.f; }

  __shared__ __align__(16) bf16_t pshm[16 * 32];
  const float scale = rsqrtf((float)cDQ);
  int ktmax = (q0 + 15) >> 5;

  for (int kt = 0; kt <= ktmax; ++kt) {
    int kb = kt * 32;
    v8f s0 = {}, s1 = {};
#pragma unroll
    for (int c = 0; c < 6; ++c) {
      const bf16_t* p0 = Kf + (size_t)(kb + n) * cDQ + c * 32 + 16 * sub;
      const bf16_t* p1 = Kf + (size_t)(kb + 16 + n) * cDQ + c * 32 + 16 * sub;
      s0 = wmma_bf16(qa[c], frag16(ld8(p0), ld8(p0 + 8)), s0);
      s1 = wmma_bf16(qa[c], frag16(ld8(p1), ld8(p1 + 8)), s1);
    }
#pragma unroll
    for (int r = 0; r < 8; ++r) {
      int qrow = q0 + r + 8 * sub;
      float a0 = s0[r] * scale;
      float a1 = s1[r] * scale;
      if (kb + n > qrow) a0 = -1e30f;
      if (kb + 16 + n > qrow) a1 = -1e30f;
      float mx = fmaxf(a0, a1);
#pragma unroll
      for (int off = 8; off; off >>= 1) mx = fmaxf(mx, __shfl_xor(mx, off, 16));
      float mnew = fmaxf(rm[r], mx);
      float corr = __expf(rm[r] - mnew);
      float p0 = __expf(a0 - mnew);
      float p1 = __expf(a1 - mnew);
      float psum = p0 + p1;
#pragma unroll
      for (int off = 8; off; off >>= 1) psum += __shfl_xor(psum, off, 16);
      rs[r] = rs[r] * corr + psum;
      rm[r] = mnew;
#pragma unroll
      for (int c2 = 0; c2 < 8; ++c2) o[c2][r] = o[c2][r] * corr;
      pshm[(r + 8 * sub) * 32 + n] = (bf16_t)p0;
      pshm[(r + 8 * sub) * 32 + 16 + n] = (bf16_t)p1;
    }
    __syncthreads();
    const bf16_t* pp = pshm + n * 32 + sub * 16;
    v16bf pa = frag16(ld8(pp), ld8(pp + 8));
#pragma unroll
    for (int c2 = 0; c2 < 8; ++c2) {
      const bf16_t* vp = VT + (size_t)(c2 * 16 + n) * cS + kb + 16 * sub;
      o[c2] = wmma_bf16(pa, frag16(ld8(vp), ld8(vp + 8)), o[c2]);
    }
    __syncthreads();
  }
#pragma unroll
  for (int r = 0; r < 8; ++r) {
    int tok = b * cS + q0 + r + 8 * sub;
    float inv = 1.f / rs[r];
#pragma unroll
    for (int c2 = 0; c2 < 8; ++c2)
      ao[(size_t)tok * (cNH * cDV) + h * cDV + c2 * 16 + n] = (bf16_t)(o[c2][r] * inv);
  }
}

// ---------------- MoE gating + scatter -------------------------------------

__global__ void gating_kernel(const float* __restrict__ h2f, const float* __restrict__ gate_w,
                              const float* __restrict__ gate_bias, int* __restrict__ ecnt,
                              int* __restrict__ elist, float* __restrict__ ewt) {
  int t = blockIdx.x;
  int lane = threadIdx.x;
  __shared__ float sc_s[cE], scb_s[cE];
  if (lane < cE) {
    const float* xr = h2f + (size_t)t * cHID;
    const float* gr = gate_w + (size_t)lane * cHID;
    float acc = 0.f;
    for (int d = 0; d < cHID; ++d) acc = fmaf(xr[d], gr[d], acc);
    float s = 1.f / (1.f + __expf(-acc));
    sc_s[lane] = s;
    scb_s[lane] = s + gate_bias[lane];
  }
  __syncthreads();
  if (lane == 0) {
    float gsc[8];
#pragma unroll
    for (int g = 0; g < 8; ++g) gsc[g] = scb_s[2 * g] + scb_s[2 * g + 1];
    bool sel[8] = {};
    for (int it = 0; it < 4; ++it) {
      int best = 0; float bv = -3.4e38f;
      for (int g = 0; g < 8; ++g)
        if (!sel[g] && gsc[g] > bv) { bv = gsc[g]; best = g; }
      sel[best] = true;
    }
    int idxs[8]; int cnt = 0;
    for (int g = 0; g < 8; ++g)
      if (sel[g]) { idxs[cnt++] = 2 * g; idxs[cnt++] = 2 * g + 1; }
    float wsum = 0.f, wv[8];
    for (int i = 0; i < 8; ++i) { wv[i] = sc_s[idxs[i]]; wsum += wv[i]; }
    float norm = cRSF / (wsum + 1e-20f);
    for (int i = 0; i < 8; ++i) {
      int e = idxs[i];
      int pos = atomicAdd(&ecnt[e], 1);
      elist[e * cT + pos] = t;
      ewt[e * cT + pos] = wv[i] * norm;
    }
  }
}

// ---------------- MoE gate+up (gathered rows) + SiLU epilogue --------------
// wgT/wuT = [E][IM][HID] bf16. Block 8 waves (2x4), tile 64 tokens x 64 IM;
// wave = 32 rows x 16 cols, 4 accumulators (gate lo/hi, up lo/hi).

__global__ void moe_gateup(const bf16_t* __restrict__ h2b, const bf16_t* __restrict__ wgT,
                           const bf16_t* __restrict__ wuT, const int* __restrict__ ecnt,
                           const int* __restrict__ elist, bf16_t* __restrict__ act) {
  int e = blockIdx.z;
  int count = ecnt[e];
  int rb = blockIdx.y * 64;
  if (rb >= count) return;
  int lane = threadIdx.x & 31, wave = threadIdx.x >> 5;
  int row0 = rb + (wave >> 2) * 32;
  int col0 = blockIdx.x * 64 + (wave & 3) * 16;
  int sub = lane >> 4, n = lane & 15;
  int r0 = row0 + n, r1 = row0 + 16 + n;
  int tok0 = (r0 < count) ? elist[e * cT + r0] : 0;
  int tok1 = (r1 < count) ? elist[e * cT + r1] : 0;
  const bf16_t* A0 = h2b + (size_t)tok0 * cHID;
  const bf16_t* A1 = h2b + (size_t)tok1 * cHID;
  const bf16_t* BgT = wgT + (size_t)e * cIM * cHID;
  const bf16_t* BuT = wuT + (size_t)e * cIM * cHID;
  v8f cg0 = {}, cg1 = {}, cu0 = {}, cu1 = {};
  for (int k0 = 0; k0 < cHID; k0 += 32) {
    const bf16_t* p0 = A0 + k0 + sub * 16;
    const bf16_t* p1 = A1 + k0 + sub * 16;
    v16bf a0 = frag16(ld8(p0), ld8(p0 + 8));
    v16bf a1 = frag16(ld8(p1), ld8(p1 + 8));
    v16bf bg = load_bT_frag(BgT, cHID, k0, col0, lane);
    v16bf bu = load_bT_frag(BuT, cHID, k0, col0, lane);
    cg0 = wmma_bf16(a0, bg, cg0);
    cu0 = wmma_bf16(a0, bu, cu0);
    cg1 = wmma_bf16(a1, bg, cg1);
    cu1 = wmma_bf16(a1, bu, cu1);
  }
#pragma unroll
  for (int rr = 0; rr < 8; ++rr) {
    int rowa = row0 + rr + 8 * sub;
    int rowb = rowa + 16;
    float g0 = cg0[rr], u0 = cu0[rr];
    float g1 = cg1[rr], u1 = cu1[rr];
    act[((size_t)e * cT + rowa) * cIM + col0 + n] =
        (bf16_t)(g0 / (1.f + __expf(-g0)) * u0);
    act[((size_t)e * cT + rowb) * cIM + col0 + n] =
        (bf16_t)(g1 / (1.f + __expf(-g1)) * u1);
  }
}

// ---------------- MoE down-proj: scatter-add weighted outputs --------------
// wdT = [E][HID][IM] bf16. Block tile 64 x 128; wave 32x32.

__global__ void moe_down(const bf16_t* __restrict__ act, const bf16_t* __restrict__ wdT,
                         const int* __restrict__ ecnt, const int* __restrict__ elist,
                         const float* __restrict__ ewt, float* __restrict__ routed) {
  int e = blockIdx.z;
  int count = ecnt[e];
  int rb = blockIdx.y * 64;
  if (rb >= count) return;
  int lane = threadIdx.x & 31, wave = threadIdx.x >> 5;
  int row0 = rb + (wave >> 2) * 32;
  int col0 = blockIdx.x * 128 + (wave & 3) * 32;
  const bf16_t* A = act + (size_t)e * cT * cIM;
  const bf16_t* BT = wdT + (size_t)e * cHID * cIM;
  v8f c00 = {}, c01 = {}, c10 = {}, c11 = {};
  for (int k0 = 0; k0 < cIM; k0 += 32) {
    v16bf a0 = load_a_frag(A, cIM, row0, k0, lane);
    v16bf a1 = load_a_frag(A, cIM, row0 + 16, k0, lane);
    v16bf b0 = load_bT_frag(BT, cIM, k0, col0, lane);
    v16bf b1 = load_bT_frag(BT, cIM, k0, col0 + 16, lane);
    c00 = wmma_bf16(a0, b0, c00);
    c01 = wmma_bf16(a0, b1, c01);
    c10 = wmma_bf16(a1, b0, c10);
    c11 = wmma_bf16(a1, b1, c11);
  }
  int sub = lane >> 4, n = lane & 15;
#pragma unroll
  for (int rr = 0; rr < 8; ++rr) {
#pragma unroll
    for (int half2 = 0; half2 < 2; ++half2) {
      int row = row0 + half2 * 16 + rr + 8 * sub;
      if (row < count) {
        int tok = elist[e * cT + row];
        float w = ewt[e * cT + row];
        float va = half2 ? c10[rr] : c00[rr];
        float vb = half2 ? c11[rr] : c01[rr];
        atomicAdd(&routed[(size_t)tok * cHID + col0 + n], w * va);
        atomicAdd(&routed[(size_t)tok * cHID + col0 + 16 + n], w * vb);
      }
    }
  }
}

// ---------------- Shared expert gate+up (dense) ----------------------------

__global__ void shared_gateup(const bf16_t* __restrict__ X, const bf16_t* __restrict__ BgT,
                              const bf16_t* __restrict__ BuT, bf16_t* __restrict__ act) {
  int lane = threadIdx.x & 31, wave = threadIdx.x >> 5;
  int row0 = blockIdx.y * 64 + (wave >> 2) * 32;
  int col0 = blockIdx.x * 64 + (wave & 3) * 16;
  int sub = lane >> 4, n = lane & 15;
  v8f cg0 = {}, cg1 = {}, cu0 = {}, cu1 = {};
  for (int k0 = 0; k0 < cHID; k0 += 32) {
    v16bf a0 = load_a_frag(X, cHID, row0, k0, lane);
    v16bf a1 = load_a_frag(X, cHID, row0 + 16, k0, lane);
    v16bf bg = load_bT_frag(BgT, cHID, k0, col0, lane);
    v16bf bu = load_bT_frag(BuT, cHID, k0, col0, lane);
    cg0 = wmma_bf16(a0, bg, cg0);
    cu0 = wmma_bf16(a0, bu, cu0);
    cg1 = wmma_bf16(a1, bg, cg1);
    cu1 = wmma_bf16(a1, bu, cu1);
  }
#pragma unroll
  for (int rr = 0; rr < 8; ++rr) {
    int rowa = row0 + rr + 8 * sub;
    int rowb = rowa + 16;
    float g0 = cg0[rr], u0 = cu0[rr];
    float g1 = cg1[rr], u1 = cu1[rr];
    act[(size_t)rowa * cIM + col0 + n] = (bf16_t)(g0 / (1.f + __expf(-g0)) * u0);
    act[(size_t)rowb * cIM + col0 + n] = (bf16_t)(g1 / (1.f + __expf(-g1)) * u1);
  }
}

// ---------------- Final combine --------------------------------------------

__global__ void final_combine(const float* __restrict__ h, const float* __restrict__ routed,
                              const float* __restrict__ sdown, float* __restrict__ out,
                              int total) {
  int i = blockIdx.x * blockDim.x + threadIdx.x;
  if (i < total) out[i] = h[i] + routed[i] + sdown[i];
}

// ---------------------------------------------------------------------------

extern "C" void kernel_launch(void* const* d_in, const int* in_sizes, int n_in,
                              void* d_out, int out_size, void* d_ws, size_t ws_size,
                              hipStream_t stream) {
  (void)in_sizes; (void)n_in; (void)out_size; (void)ws_size;
  const float* x         = (const float*)d_in[0];
  const float* norm1_w   = (const float*)d_in[1];
  const float* w_q_a     = (const float*)d_in[2];
  const float* q_a_norm  = (const float*)d_in[3];
  const float* w_q_b     = (const float*)d_in[4];
  const float* w_kv_a    = (const float*)d_in[5];
  const float* kv_a_norm = (const float*)d_in[6];
  const float* w_kv_b    = (const float*)d_in[7];
  const float* w_out     = (const float*)d_in[8];
  const float* norm2_w   = (const float*)d_in[9];
  const float* gate_w    = (const float*)d_in[10];
  const float* gate_bias = (const float*)d_in[11];
  const float* w_gate    = (const float*)d_in[12];
  const float* w_up      = (const float*)d_in[13];
  const float* w_down    = (const float*)d_in[14];
  const float* ws_gate   = (const float*)d_in[15];
  const float* ws_up     = (const float*)d_in[16];
  const float* ws_down   = (const float*)d_in[17];
  float* out = (float*)d_out;

  char* wp = (char*)d_ws;
  auto alloc = [&](size_t bytes) -> void* {
    void* p = (void*)wp;
    wp += (bytes + 255) & ~(size_t)255;
    return p;
  };
  // activations
  bf16_t* h1_bf   = (bf16_t*)alloc((size_t)cT * cHID * 2);
  float* qa_f     = (float*)alloc((size_t)cT * cQR * 4);
  bf16_t* qa_bf   = (bf16_t*)alloc((size_t)cT * cQR * 2);
  float* q_f      = (float*)alloc((size_t)cT * (cNH * cDQ) * 4);
  float* kva_f    = (float*)alloc((size_t)cT * (cKVR + 64) * 4);
  bf16_t* kvl_bf  = (bf16_t*)alloc((size_t)cT * cKVR * 2);
  float* kv_f     = (float*)alloc((size_t)cT * (cNH * (cDN + cDV)) * 4);
  bf16_t* qf_bf   = (bf16_t*)alloc((size_t)cB * cNH * cS * cDQ * 2);
  bf16_t* kf_bf   = (bf16_t*)alloc((size_t)cB * cNH * cS * cDQ * 2);
  bf16_t* vt_bf   = (bf16_t*)alloc((size_t)cB * cNH * cDV * cS * 2);
  bf16_t* ao_bf   = (bf16_t*)alloc((size_t)cT * (cNH * cDV) * 2);
  float* h_f      = (float*)alloc((size_t)cT * cHID * 4);
  bf16_t* h2_bf   = (bf16_t*)alloc((size_t)cT * cHID * 2);
  float* h2_f     = (float*)alloc((size_t)cT * cHID * 4);
  int* ecnt       = (int*)alloc((size_t)cE * 4);
  int* elist      = (int*)alloc((size_t)cE * cT * 4);
  float* ewt      = (float*)alloc((size_t)cE * cT * 4);
  bf16_t* act_bf  = (bf16_t*)alloc((size_t)cE * cT * cIM * 2);
  float* routed_f = (float*)alloc((size_t)cT * cHID * 4);
  bf16_t* sact_bf = (bf16_t*)alloc((size_t)cT * cIM * 2);
  float* sdown_f  = (float*)alloc((size_t)cT * cHID * 4);
  // transposed bf16 weights
  bf16_t* wqaT  = (bf16_t*)alloc((size_t)cQR * cHID * 2);
  bf16_t* wqbT  = (bf16_t*)alloc((size_t)(cNH * cDQ) * cQR * 2);
  bf16_t* wkvaT = (bf16_t*)alloc((size_t)(cKVR + 64) * cHID * 2);
  bf16_t* wkvbT = (bf16_t*)alloc((size_t)(cNH * (cDN + cDV)) * cKVR * 2);
  bf16_t* woutT = (bf16_t*)alloc((size_t)cHID * (cNH * cDV) * 2);
  bf16_t* wgT   = (bf16_t*)alloc((size_t)cE * cIM * cHID * 2);
  bf16_t* wuT   = (bf16_t*)alloc((size_t)cE * cIM * cHID * 2);
  bf16_t* wdT   = (bf16_t*)alloc((size_t)cE * cHID * cIM * 2);
  bf16_t* wsgT  = (bf16_t*)alloc((size_t)cIM * cHID * 2);
  bf16_t* wsuT  = (bf16_t*)alloc((size_t)cIM * cHID * 2);
  bf16_t* wsdT  = (bf16_t*)alloc((size_t)cHID * cIM * 2);

  auto tconv = [&](const float* W, bf16_t* WT, int K, int N, int Z) {
    convert_transpose<<<dim3(N / 32, K / 32, Z), 256, 0, stream>>>(W, WT, K, N);
  };
  tconv(w_q_a, wqaT, cHID, cQR, 1);
  tconv(w_q_b, wqbT, cQR, cNH * cDQ, 1);
  tconv(w_kv_a, wkvaT, cHID, cKVR + 64, 1);
  tconv(w_kv_b, wkvbT, cKVR, cNH * (cDN + cDV), 1);
  tconv(w_out, woutT, cNH * cDV, cHID, 1);
  tconv(w_gate, wgT, cHID, cIM, cE);
  tconv(w_up, wuT, cHID, cIM, cE);
  tconv(w_down, wdT, cIM, cHID, cE);
  tconv(ws_gate, wsgT, cHID, cIM, 1);
  tconv(ws_up, wsuT, cHID, cIM, 1);
  tconv(ws_down, wsdT, cIM, cHID, 1);

  auto gemm = [&](const bf16_t* A, const bf16_t* BT, float* C, const float* Cin,
                  int M, int N, int K, int lda, int ldc) {
    gemm_wmma<<<dim3((N + 127) / 128, (M + 63) / 64), 256, 0, stream>>>(A, BT, C, Cin,
                                                                        M, N, K, lda, ldc);
  };

  // 1) h1 = rms(x)*norm1_w
  rmsnorm_kernel<<<cT, 256, 0, stream>>>(x, norm1_w, h1_bf, nullptr, cHID, cHID);
  // 2) q path
  gemm(h1_bf, wqaT, qa_f, nullptr, cT, cQR, cHID, cHID, cQR);
  rmsnorm_kernel<<<cT, 256, 0, stream>>>(qa_f, q_a_norm, qa_bf, nullptr, cQR, cQR);
  gemm(qa_bf, wqbT, q_f, nullptr, cT, cNH * cDQ, cQR, cQR, cNH * cDQ);
  // 3) kv path
  gemm(h1_bf, wkvaT, kva_f, nullptr, cT, cKVR + 64, cHID, cHID, cKVR + 64);
  rmsnorm_kernel<<<cT, 256, 0, stream>>>(kva_f, kv_a_norm, kvl_bf, nullptr, cKVR, cKVR + 64);
  gemm(kvl_bf, wkvbT, kv_f, nullptr, cT, cNH * (cDN + cDV), cKVR, cKVR, cNH * (cDN + cDV));
  // 4) reshape + RoPE
  qkv_prep<<<dim3(cS, cNH, cB), 192, 0, stream>>>(q_f, kv_f, kva_f, qf_bf, kf_bf, vt_bf);
  // 5) causal flash attention
  attn_kernel<<<dim3(cS / 16, cNH, cB), 32, 0, stream>>>(qf_bf, kf_bf, vt_bf, ao_bf);
  // 6) h = x + ao @ w_out
  gemm(ao_bf, woutT, h_f, x, cT, cHID, cNH * cDV, cNH * cDV, cHID);
  // 7) h2 = rms(h)
  rmsnorm_kernel<<<cT, 256, 0, stream>>>(h_f, norm2_w, h2_bf, h2_f, cHID, cHID);
  // 8) gating + scatter
  hipMemsetAsync(ecnt, 0, cE * sizeof(int), stream);
  hipMemsetAsync(routed_f, 0, (size_t)cT * cHID * 4, stream);
  gating_kernel<<<cT, 32, 0, stream>>>(h2_f, gate_w, gate_bias, ecnt, elist, ewt);
  // 9) routed experts
  moe_gateup<<<dim3(cIM / 64, cT / 64, cE), 256, 0, stream>>>(h2_bf, wgT, wuT, ecnt,
                                                              elist, act_bf);
  moe_down<<<dim3(cHID / 128, cT / 64, cE), 256, 0, stream>>>(act_bf, wdT, ecnt, elist,
                                                              ewt, routed_f);
  // 10) shared expert
  shared_gateup<<<dim3(cIM / 64, cT / 64), 256, 0, stream>>>(h2_bf, wsgT, wsuT, sact_bf);
  gemm(sact_bf, wsdT, sdown_f, nullptr, cT, cHID, cIM, cIM, cHID);
  // 11) out = h + routed + shared
  final_combine<<<(cT * cHID + 255) / 256, 256, 0, stream>>>(h_f, routed_f, sdown_f, out,
                                                             cT * cHID);
}
